// GPT_14164802142920
// MI455X (gfx1250) — matrix operand order repaired and checked by reading, not compiled, not run on hardware
//
#include <hip/hip_runtime.h>
#include <hip/hip_bf16.h>
#include <math.h>

// ---- model constants (fixed by the reference) ----
#define BB   8
#define TT   1024
#define DD   1024
#define HH   16
#define LL   8
#define VV   8192
#define HD   64
#define MM   (BB * TT)          // 8192 rows
#define SCALE_ATT 0.125f        // 1/sqrt(64)

typedef __bf16 bf16;
typedef __attribute__((ext_vector_type(16))) bf16  v16bf;
typedef __attribute__((ext_vector_type(8)))  bf16  v8bf;
typedef __attribute__((ext_vector_type(8)))  float v8f;
typedef __attribute__((ext_vector_type(16))) float v16f;

static __device__ __forceinline__ v8f wmma_bf16(v16bf a, v16bf b, v8f c) {
    // D = A(16x32 bf16) * B(32x16 bf16) + C(16x16 f32)
    return __builtin_amdgcn_wmma_f32_16x16x32_bf16(
        false, a, false, b, (short)0, c, false, false);
}

// B-fragment loaders (16 contiguous K values per lane)
static __device__ __forceinline__ v16bf load_bfrag(const bf16* p) {
    return *(const v16bf*)p;                       // 32B: two b128 loads
}
static __device__ __forceinline__ v16bf load_bfrag(const float* p) {
    v16f wv = *(const v16f*)p;                     // 64B fp32 + convert
    union { v16bf v; bf16 e[16]; } ub;
#pragma unroll
    for (int e = 0; e < 16; ++e) ub.e[e] = (bf16)wv[e];
    return ub.v;
}

// =====================================================================
// fp32 -> bf16 bulk weight conversion (8 elems/thread, vectorized)
// =====================================================================
__global__ __launch_bounds__(256) void cvt_kernel(
    const float* __restrict__ in, bf16* __restrict__ out, int n)
{
    int i = ((int)blockIdx.x * 256 + (int)threadIdx.x) * 8;
    if (i >= n) return;
    v8f v = *(const v8f*)(in + i);
    union { v8bf b; bf16 e[8]; } u;
#pragma unroll
    for (int e = 0; e < 8; ++e) u.e[e] = (bf16)v[e];
    *(v8bf*)(out + i) = u.b;
}

// =====================================================================
// Embedding
// =====================================================================
__global__ __launch_bounds__(256) void embed_kernel(
    const int* __restrict__ idx, const int* __restrict__ ts,
    const float* __restrict__ tok, const float* __restrict__ pos,
    const float* __restrict__ gpe, float* __restrict__ x)
{
    size_t i  = (size_t)blockIdx.x * 256 + threadIdx.x;   // over MM*DD
    int    d  = (int)(i & (DD - 1));
    size_t bt = i >> 10;
    int    t  = (int)(bt & (TT - 1));
    int    b  = (int)(bt >> 10);
    x[i] = tok[(size_t)idx[bt] * DD + d]
         + gpe[(size_t)ts[b] * DD + d]
         + pos[(size_t)t * DD + d];
}

// =====================================================================
// LayerNorm (fp32 in) -> bf16 out; one 256-thread block per row of D=1024
// =====================================================================
__global__ __launch_bounds__(256) void ln_kernel(
    const float* __restrict__ x, const float* __restrict__ w,
    const float* __restrict__ b, bf16* __restrict__ out)
{
    __shared__ float s1[256], s2[256];
    int row = blockIdx.x, tid = threadIdx.x;
    const float* xr = x + (size_t)row * DD;
    float vals[4], s = 0.f, q = 0.f;
#pragma unroll
    for (int i = 0; i < 4; ++i) {
        float v = xr[tid + i * 256];
        vals[i] = v; s += v; q += v * v;
    }
    s1[tid] = s; s2[tid] = q; __syncthreads();
    for (int off = 128; off > 0; off >>= 1) {
        if (tid < off) { s1[tid] += s1[tid + off]; s2[tid] += s2[tid + off]; }
        __syncthreads();
    }
    float mean = s1[0] * (1.f / DD);
    float var  = s2[0] * (1.f / DD) - mean * mean;
    float rstd = rsqrtf(var + 1e-5f);
#pragma unroll
    for (int i = 0; i < 4; ++i) {
        int d = tid + i * 256;
        out[(size_t)row * DD + d] = (bf16)((vals[i] - mean) * rstd * w[d] + b[d]);
    }
}

// =====================================================================
// WMMA GEMM: C[m,n] = sum_k A[m,k] * W[n,k] (+ bias[n])
//   A: bf16 [M x K]; W: [N x K] bf16 (preconverted) or fp32 (fallback).
//   Block 128x128, 8 waves (4x2); wave = 2 mtiles x 4 ntiles; K step 32.
// Epilogues: 0 f32 store | 1 f32 accumulate | 2 bf16 [B,H,T,64]
//            3 bf16 [B,H,64,T] | 4 GELU -> bf16
// =====================================================================
template <typename WT>
__global__ __launch_bounds__(256) void gemm_kernel(
    const bf16* __restrict__ A, const WT* __restrict__ W,
    const float* __restrict__ bias, void* __restrict__ out,
    int M, int N, int K, int epi)
{
    const int lane = threadIdx.x & 31;
    const int wid  = threadIdx.x >> 5;
    const int wm   = wid >> 1, wn = wid & 1;
    const int l15  = lane & 15;
    const int half = lane >> 4;
    const int mbase = blockIdx.x * 128 + wm * 32;   // 2 mtiles
    const int nbase = blockIdx.y * 128 + wn * 64;   // 4 ntiles

    const v8f zero8 = {0.f,0.f,0.f,0.f,0.f,0.f,0.f,0.f};
    v8f acc[2][4];
#pragma unroll
    for (int i = 0; i < 2; ++i)
#pragma unroll
        for (int j = 0; j < 4; ++j) acc[i][j] = zero8;

#pragma unroll 2
    for (int k0 = 0; k0 < K; k0 += 32) {
        v16bf afr[2], bfr[4];
#pragma unroll
        for (int mt = 0; mt < 2; ++mt) {
            // A 16x32 frag, this code's contiguous K order (B matches): lane<16 -> K 0..15 ; lane>=16 -> K 16..31
            const bf16* pa = A + (size_t)(mbase + mt * 16 + l15) * K + k0 + half * 16;
            union { v16bf v; v8bf h[2]; } ua;
            ua.h[0] = *(const v8bf*)(pa);
            ua.h[1] = *(const v8bf*)(pa + 8);
            afr[mt] = ua.v;
            __builtin_prefetch(pa + 64, 0, 1);     // 2 K-steps ahead
        }
#pragma unroll
        for (int nt = 0; nt < 4; ++nt) {
            const WT* pw = W + (size_t)(nbase + nt * 16 + l15) * K + k0 + half * 16;
            bfr[nt] = load_bfrag(pw);
            __builtin_prefetch(pw + 64, 0, 1);     // 2 K-steps ahead
        }
#pragma unroll
        for (int mt = 0; mt < 2; ++mt)
#pragma unroll
            for (int nt = 0; nt < 4; ++nt)
                acc[mt][nt] = wmma_bf16(afr[mt], bfr[nt], acc[mt][nt]);
    }

#pragma unroll
    for (int mt = 0; mt < 2; ++mt) {
#pragma unroll
        for (int nt = 0; nt < 4; ++nt) {
#pragma unroll
            for (int r = 0; r < 8; ++r) {
                int m = mbase + mt * 16 + r + half * 8;
                int n = nbase + nt * 16 + l15;
                float v = acc[mt][nt][r];
                if (bias) v += bias[n];
                if (epi == 0) {
                    ((float*)out)[(size_t)m * N + n] = v;
                } else if (epi == 1) {
                    float* o = (float*)out;
                    o[(size_t)m * N + n] += v;
                } else if (epi == 2) {
                    int b = m >> 10, t = m & (TT - 1);
                    int h = n >> 6,  d = n & (HD - 1);
                    ((bf16*)out)[(((size_t)(b * HH + h)) * TT + t) * HD + d] = (bf16)v;
                } else if (epi == 3) {
                    int b = m >> 10, t = m & (TT - 1);
                    int h = n >> 6,  d = n & (HD - 1);
                    ((bf16*)out)[(((size_t)(b * HH + h)) * HD + d) * TT + t] = (bf16)v;
                } else { // 4: exact GELU
                    float g = 0.5f * v * (1.f + erff(v * 0.70710678f));
                    ((bf16*)out)[(size_t)m * N + n] = (bf16)g;
                }
            }
        }
    }
}

// =====================================================================
// Flash attention: one WG per (b, h, 64-row q block). 256 threads, 8 waves.
//   Qh, Kh: bf16 [B,H,T,64] ; Vt: bf16 [B,H,64,T] ; out ybuf: bf16 [B*T, D]
// =====================================================================
__global__ __launch_bounds__(256) void attn_kernel(
    const bf16* __restrict__ Qh, const bf16* __restrict__ Kh,
    const bf16* __restrict__ Vt, bf16* __restrict__ ybuf)
{
    __shared__ float S[64][128];     // 32 KB
    __shared__ bf16  P[64][128];     // 16 KB
    __shared__ float pbuf[4][64];    // softmax partial reduce
    __shared__ float rmax[64], rsum[64], rcorr[64];

    const int qb = blockIdx.x & 15;          // T/64 = 16
    const int bh = blockIdx.x >> 4;
    const int b  = bh >> 4, h = bh & 15;     // H = 16
    const int q0 = qb * 64;

    const bf16* Q  = Qh + (size_t)bh * TT * HD;
    const bf16* Kp = Kh + (size_t)bh * TT * HD;
    const bf16* Vp = Vt + (size_t)bh * HD * TT;

    const int tid  = threadIdx.x;
    const int lane = tid & 31, wid = tid >> 5;
    const int l15  = lane & 15, half = lane >> 4;
    const int mt   = wid >> 1;               // this wave's M tile (both GEMM phases)
    const int srow = tid & 63, ssub = tid >> 6;  // softmax: 4 threads per row

    if (tid < 64) { rmax[tid] = -1e30f; rsum[tid] = 0.f; }

    // Q fragments are invariant over the chunk loop: load once.
    v16bf aQ[2];
#pragma unroll
    for (int ks = 0; ks < 2; ++ks) {
        const bf16* pq = Q + (size_t)(q0 + mt * 16 + l15) * HD + ks * 32 + half * 16;
        union { v16bf v; v8bf hh[2]; } ua;
        ua.hh[0] = *(const v8bf*)(pq);
        ua.hh[1] = *(const v8bf*)(pq + 8);
        aQ[ks] = ua.v;
    }

    const v8f zero8 = {0.f,0.f,0.f,0.f,0.f,0.f,0.f,0.f};
    v8f yacc[2]; yacc[0] = zero8; yacc[1] = zero8;

    const int kend = q0 + 64;                // causal upper bound
    for (int kc = 0; kc < kend; kc += 128) {
        const int clen = min(128, kend - kc);   // 64 or 128 (uniform)

        // ---- phase 1: scores S[64][128] ----
#pragma unroll
        for (int ntl = 0; ntl < 4; ++ntl) {
            int nt = (wid & 1) * 4 + ntl;
            v8f sacc = zero8;
            if (nt * 16 < clen) {               // wave-uniform at runtime
                int keyn = kc + nt * 16 + l15;
#pragma unroll
                for (int ks = 0; ks < 2; ++ks) {
                    const bf16* pk = Kp + (size_t)keyn * HD + ks * 32 + half * 16;
                    v16bf bK = *(const v16bf*)pk;
                    sacc = wmma_bf16(aQ[ks], bK, sacc);
                }
            }
#pragma unroll
            for (int r = 0; r < 8; ++r) {
                int row = mt * 16 + r + half * 8;
                int col = nt * 16 + l15;
                float sv = sacc[r] * SCALE_ATT;
                if ((q0 + row < kc + col) || (col >= clen)) sv = -1e30f;
                S[row][col] = sv;
            }
        }
        __syncthreads();

        // ---- phase 2: online softmax, 4 threads per row ----
        {
            int j0 = ssub * 32;
            float mx = -1e30f;
            int jm = min(clen, j0 + 32);
            for (int j = j0; j < jm; ++j) mx = fmaxf(mx, S[srow][j]);
            pbuf[ssub][srow] = mx;
            __syncthreads();
            float m0 = rmax[srow];
            float nm = fmaxf(m0, fmaxf(fmaxf(pbuf[0][srow], pbuf[1][srow]),
                                       fmaxf(pbuf[2][srow], pbuf[3][srow])));
            float s = 0.f;
            for (int j = j0; j < j0 + 32; ++j) {
                float p = (j < clen) ? __expf(S[srow][j] - nm) : 0.f;
                P[srow][j] = (bf16)p;
                s += p;
            }
            __syncthreads();                 // pbuf max values consumed
            pbuf[ssub][srow] = s;
            __syncthreads();
            if (ssub == 0) {
                float corr = __expf(m0 - nm);
                rsum[srow] = rsum[srow] * corr
                           + pbuf[0][srow] + pbuf[1][srow] + pbuf[2][srow] + pbuf[3][srow];
                rmax[srow] = nm;
                rcorr[srow] = corr;
            }
            __syncthreads();
        }

        // ---- phase 3: rescale accumulators, Y += P @ V ----
#pragma unroll
        for (int j = 0; j < 2; ++j)
#pragma unroll
            for (int r = 0; r < 8; ++r)
                yacc[j][r] *= rcorr[mt * 16 + r + half * 8];

        const int ksteps = clen >> 5;           // 2 or 4 (uniform)
        for (int ks = 0; ks < ksteps; ++ks) {
            const bf16* pp = &P[mt * 16 + l15][ks * 32 + half * 16];
            union { v16bf v; v8bf hh[2]; } ua;
            ua.hh[0] = *(const v8bf*)(pp);
            ua.hh[1] = *(const v8bf*)(pp + 8);
#pragma unroll
            for (int j = 0; j < 2; ++j) {
                int nt = (wid & 1) * 2 + j;
                const bf16* pv = Vp + (size_t)(nt * 16 + l15) * TT + kc + ks * 32 + half * 16;
                v16bf bV = *(const v16bf*)pv;
                yacc[j] = wmma_bf16(ua.v, bV, yacc[j]);
            }
        }
        __syncthreads();
    }

    // ---- epilogue: Y / rowsum -> ybuf [B*T, D] bf16 ----
#pragma unroll
    for (int j = 0; j < 2; ++j) {
        int nt = (wid & 1) * 2 + j;
#pragma unroll
        for (int r = 0; r < 8; ++r) {
            int row = mt * 16 + r + half * 8;
            float y = yacc[j][r] / rsum[row];
            size_t grow = (size_t)b * TT + q0 + row;
            int col = h * HD + nt * 16 + l15;
            ybuf[grow * DD + col] = (bf16)y;
        }
    }
}

// =====================================================================
// Host orchestration
// =====================================================================
extern "C" void kernel_launch(void* const* d_in, const int* in_sizes, int n_in,
                              void* d_out, int out_size, void* d_ws, size_t ws_size,
                              hipStream_t stream) {
    (void)in_sizes; (void)n_in; (void)out_size;
    const int*   idx   = (const int*)  d_in[0];
    const int*   ts    = (const int*)  d_in[1];
    const float* tok   = (const float*)d_in[2];
    const float* pos   = (const float*)d_in[3];
    const float* gpe   = (const float*)d_in[4];
    const float* ln1w  = (const float*)d_in[5];
    const float* ln1b  = (const float*)d_in[6];
    const float* Wq    = (const float*)d_in[7];
    const float* bq    = (const float*)d_in[8];
    const float* Wk    = (const float*)d_in[9];
    const float* bk    = (const float*)d_in[10];
    const float* Wv    = (const float*)d_in[11];
    const float* bv    = (const float*)d_in[12];
    const float* Wo    = (const float*)d_in[13];
    const float* bo    = (const float*)d_in[14];
    const float* ln2w  = (const float*)d_in[15];
    const float* ln2b  = (const float*)d_in[16];
    const float* W1    = (const float*)d_in[17];
    const float* b1    = (const float*)d_in[18];
    const float* W2    = (const float*)d_in[19];
    const float* b2    = (const float*)d_in[20];
    const float* lnfw  = (const float*)d_in[21];
    const float* lnfb  = (const float*)d_in[22];
    const float* headw = (const float*)d_in[23];

    // activation workspace
    char* w = (char*)d_ws;
    float* x   = (float*)w; w += (size_t)MM * DD * sizeof(float);
    bf16* hbuf = (bf16*)w;  w += (size_t)MM * DD * sizeof(bf16);
    bf16* Qh   = (bf16*)w;  w += (size_t)MM * DD * sizeof(bf16);
    bf16* Kh   = (bf16*)w;  w += (size_t)MM * DD * sizeof(bf16);
    bf16* Vt   = (bf16*)w;  w += (size_t)MM * DD * sizeof(bf16);
    bf16* ybuf = (bf16*)w;  w += (size_t)MM * DD * sizeof(bf16);
    bf16* hid  = (bf16*)w;  w += (size_t)MM * 2 * DD * sizeof(bf16);
    size_t actBytes = (size_t)(w - (char*)d_ws);

    // bf16 weight cache (preferred path)
    const size_t nQ = (size_t)LL * DD * DD;       // 8M each (Wq,Wk,Wv,Wo)
    const size_t n1 = (size_t)LL * 2 * DD * DD;   // 16M (W1), 16M (W2)
    const size_t nH = (size_t)VV * DD;            // 8M (head)
    size_t wElems = 4 * nQ + 2 * n1 + nH;         // 72M
    bool useBf16W = ws_size >= actBytes + wElems * sizeof(bf16);

    embed_kernel<<<(MM * DD) / 256, 256, 0, stream>>>(idx, ts, tok, pos, gpe, x);

    dim3 gD(MM / 128, DD / 128);
    dim3 g1(MM / 128, (2 * DD) / 128);
    dim3 gH(MM / 128, VV / 128);
    const int attnGrid = BB * HH * (TT / 64);

    if (useBf16W) {
        bf16* cWq = (bf16*)w;           w += nQ * sizeof(bf16);
        bf16* cWk = (bf16*)w;           w += nQ * sizeof(bf16);
        bf16* cWv = (bf16*)w;           w += nQ * sizeof(bf16);
        bf16* cWo = (bf16*)w;           w += nQ * sizeof(bf16);
        bf16* cW1 = (bf16*)w;           w += n1 * sizeof(bf16);
        bf16* cW2 = (bf16*)w;           w += n1 * sizeof(bf16);
        bf16* cWh = (bf16*)w;           w += nH * sizeof(bf16);

        cvt_kernel<<<(int)(nQ / (256 * 8)), 256, 0, stream>>>(Wq, cWq, (int)nQ);
        cvt_kernel<<<(int)(nQ / (256 * 8)), 256, 0, stream>>>(Wk, cWk, (int)nQ);
        cvt_kernel<<<(int)(nQ / (256 * 8)), 256, 0, stream>>>(Wv, cWv, (int)nQ);
        cvt_kernel<<<(int)(nQ / (256 * 8)), 256, 0, stream>>>(Wo, cWo, (int)nQ);
        cvt_kernel<<<(int)(n1 / (256 * 8)), 256, 0, stream>>>(W1, cW1, (int)n1);
        cvt_kernel<<<(int)(n1 / (256 * 8)), 256, 0, stream>>>(W2, cW2, (int)n1);
        cvt_kernel<<<(int)(nH / (256 * 8)), 256, 0, stream>>>(headw, cWh, (int)nH);

        for (int l = 0; l < LL; ++l) {
            ln_kernel<<<MM, 256, 0, stream>>>(x, ln1w + (size_t)l * DD, ln1b + (size_t)l * DD, hbuf);
            gemm_kernel<bf16><<<gD, 256, 0, stream>>>(hbuf, cWq + (size_t)l * DD * DD, bq + (size_t)l * DD, Qh, MM, DD, DD, 2);
            gemm_kernel<bf16><<<gD, 256, 0, stream>>>(hbuf, cWk + (size_t)l * DD * DD, bk + (size_t)l * DD, Kh, MM, DD, DD, 2);
            gemm_kernel<bf16><<<gD, 256, 0, stream>>>(hbuf, cWv + (size_t)l * DD * DD, bv + (size_t)l * DD, Vt, MM, DD, DD, 3);
            attn_kernel<<<attnGrid, 256, 0, stream>>>(Qh, Kh, Vt, ybuf);
            gemm_kernel<bf16><<<gD, 256, 0, stream>>>(ybuf, cWo + (size_t)l * DD * DD, bo + (size_t)l * DD, x, MM, DD, DD, 1);
            ln_kernel<<<MM, 256, 0, stream>>>(x, ln2w + (size_t)l * DD, ln2b + (size_t)l * DD, hbuf);
            gemm_kernel<bf16><<<g1, 256, 0, stream>>>(hbuf, cW1 + (size_t)l * 2 * DD * DD, b1 + (size_t)l * 2 * DD, hid, MM, 2 * DD, DD, 4);
            gemm_kernel<bf16><<<gD, 256, 0, stream>>>(hid, cW2 + (size_t)l * 2 * DD * DD, b2 + (size_t)l * DD, x, MM, DD, 2 * DD, 1);
        }
        ln_kernel<<<MM, 256, 0, stream>>>(x, lnfw, lnfb, hbuf);
        gemm_kernel<bf16><<<gH, 256, 0, stream>>>(hbuf, cWh, nullptr, d_out, MM, VV, DD, 0);
    } else {
        for (int l = 0; l < LL; ++l) {
            ln_kernel<<<MM, 256, 0, stream>>>(x, ln1w + (size_t)l * DD, ln1b + (size_t)l * DD, hbuf);
            gemm_kernel<float><<<gD, 256, 0, stream>>>(hbuf, Wq + (size_t)l * DD * DD, bq + (size_t)l * DD, Qh, MM, DD, DD, 2);
            gemm_kernel<float><<<gD, 256, 0, stream>>>(hbuf, Wk + (size_t)l * DD * DD, bk + (size_t)l * DD, Kh, MM, DD, DD, 2);
            gemm_kernel<float><<<gD, 256, 0, stream>>>(hbuf, Wv + (size_t)l * DD * DD, bv + (size_t)l * DD, Vt, MM, DD, DD, 3);
            attn_kernel<<<attnGrid, 256, 0, stream>>>(Qh, Kh, Vt, ybuf);
            gemm_kernel<float><<<gD, 256, 0, stream>>>(ybuf, Wo + (size_t)l * DD * DD, bo + (size_t)l * DD, x, MM, DD, DD, 1);
            ln_kernel<<<MM, 256, 0, stream>>>(x, ln2w + (size_t)l * DD, ln2b + (size_t)l * DD, hbuf);
            gemm_kernel<float><<<g1, 256, 0, stream>>>(hbuf, W1 + (size_t)l * 2 * DD * DD, b1 + (size_t)l * 2 * DD, hid, MM, 2 * DD, DD, 4);
            gemm_kernel<float><<<gD, 256, 0, stream>>>(hid, W2 + (size_t)l * 2 * DD * DD, b2 + (size_t)l * DD, x, MM, DD, 2 * DD, 1);
        }
        ln_kernel<<<MM, 256, 0, stream>>>(x, lnfw, lnfb, hbuf);
        gemm_kernel<float><<<gH, 256, 0, stream>>>(hbuf, headw, nullptr, d_out, MM, VV, DD, 0);
    }
}
